// Encoder_25847113188087
// MI455X (gfx1250) — matrix operand order repaired and checked by reading, not compiled, not run on hardware
//
#include <hip/hip_runtime.h>
#include <math.h>

typedef _Float16 v16h __attribute__((ext_vector_type(16)));
typedef _Float16 v8h  __attribute__((ext_vector_type(8)));
typedef float    v8f  __attribute__((ext_vector_type(8)));

#define MROWS 4096          // BS * NMASK tokens
#define CDIM  256
#define NMASK 1024
#define BSZ   4
#define TK    32            // GEMM k-step

// ---------------------------------------------------------------------------
// activation
__device__ __forceinline__ float act_apply(float x, int act) {
  switch (act) {
    case 1: return 0.5f * x * (1.0f + erff(x * 0.70710678118654752f)); // exact GELU
    case 2: return x > 0.0f ? x : 0.0f;                                // ReLU
    case 3: return 1.0f / (1.0f + __expf(-x));                         // sigmoid
    default: return x;
  }
}

// ---------------------------------------------------------------------------
// weight convert: W f32 [K,N] -> WT f16 [N,K]
__global__ void wconv_kernel(const float* __restrict__ W, _Float16* __restrict__ WT,
                             int K, int N) {
  int t = blockIdx.x * 256 + threadIdx.x;
  if (t >= K * N) return;
  int k = t / N, n = t - k * N;
  WT[(size_t)n * K + k] = (_Float16)W[t];
}

// ---------------------------------------------------------------------------
// patch embed (conv P=2) + sincos pos + cls + mask gather, fused.
__global__ void patch_gather_kernel(const float* __restrict__ x, const int* __restrict__ midx,
                                    const float* __restrict__ Wp, const float* __restrict__ bp,
                                    const float* __restrict__ cls, float* __restrict__ m0) {
  int e = blockIdx.x * 256 + threadIdx.x;
  if (e >= BSZ * NMASK * CDIM) return;
  int c = e & 255;
  int i = (e >> 8) & 1023;
  int b = e >> 18;
  int tok = midx[b * NMASK + i];
  float val;
  if (tok == 0) {
    val = cls[c];
  } else {
    int p  = tok - 1;
    int ph = p >> 6;         // H/P = 64
    int pw = p & 63;
    float acc = bp[c];
#pragma unroll
    for (int ch = 0; ch < 3; ch++)
#pragma unroll
      for (int dy = 0; dy < 2; dy++)
#pragma unroll
        for (int dx = 0; dx < 2; dx++)
          acc += x[(((size_t)b * 3 + ch) * 128 + (ph * 2 + dy)) * 128 + (pw * 2 + dx)] *
                 Wp[((c * 3 + ch) * 2 + dy) * 2 + dx];
    // 2D sincos pos embed: [sin(w),cos(w),sin(h),cos(h)] x 64 freqs each
    int cm = c & 63, grp = c >> 6;
    float posv = (grp < 2) ? (float)pw : (float)ph;
    float ang  = posv * __expf(-(float)cm * 0.14391156822f); // ln(1e4)/64
    acc += (grp & 1) ? cosf(ang) : sinf(ang);
    val = acc;
  }
  m0[e] = val;
}

// ---------------------------------------------------------------------------
// A fragment from LDS tile row: f32 -> f16 A-layout halves (ISA 16-bit A 16x32).
// p points at tileRow + 8*hi ; groups at +0..7 and +16..23.
__device__ __forceinline__ v16h load_afrag_lds(const float* __restrict__ p) {
  v16h a;
#pragma unroll
  for (int q = 0; q < 8; q++) { a[q] = (_Float16)p[q]; a[8 + q] = (_Float16)p[16 + q]; }
  return a;
}

// ---------------------------------------------------------------------------
// WMMA GEMM: C[4096,N] = act(A_f32[4096,K] @ B + bias) + residual
// BT f16 [N,K] (pre-transposed). Requires K%32==0, N%128==0, M%128==0.
// Block = 8 waves = 128x128 tile; wave = 32x64 strip (8 WMMAs per k-step).
// A tile (128x32 f32, 16KB) staged to LDS with async global->LDS copies,
// double-buffered on ASYNCcnt so the copy engine runs ahead of the WMMAs.
__global__ void gemm_f16w(const float* __restrict__ A, const _Float16* __restrict__ BT,
                          const float* __restrict__ bias, const float* __restrict__ residual,
                          float* __restrict__ Cout, int N, int K, int act) {
  __shared__ float atile[2][128][TK];                  // 2 x 16KB
  const int tid  = threadIdx.x;
  const int lane = tid & 31;
  const int w    = tid >> 5;
  const int hi   = lane >> 4;
  const int lp   = lane & 15;
  const int mblk = blockIdx.y * 128;
  const int m0   = mblk + (w >> 1) * 32;               // wave: 32 rows
  const int n0   = blockIdx.x * 128 + (w & 1) * 64;    // wave: 64 cols

  // async-stage one 128x32 k-slab of A into atile[buf]; 4 x 16B per thread.
  auto stage = [&](int buf, int ks) {
#pragma unroll
    for (int c = 0; c < 4; c++) {
      const int idx = c * 256 + tid;                   // 0..1023 16B-chunks
      const int row = idx >> 3;
      const int c4  = (idx & 7) * 4;
      const unsigned lptr = (unsigned)(size_t)&atile[buf][row][c4];
      const unsigned long long gptr =
          (unsigned long long)(size_t)(A + (size_t)(mblk + row) * K + ks + c4);
      asm volatile("global_load_async_to_lds_b128 %0, %1, off"
                   :: "v"(lptr), "v"(gptr) : "memory");
    }
  };

  v8f acc[2][4];
#pragma unroll
  for (int mi = 0; mi < 2; mi++)
#pragma unroll
    for (int t = 0; t < 4; t++)
#pragma unroll
      for (int r = 0; r < 8; r++) acc[mi][t][r] = 0.0f;

  const _Float16* bbase = BT + (size_t)(n0 + lp) * K + 8 * hi;

  stage(0, 0);
  int buf = 0;
  for (int ks = 0; ks < K; ks += TK) {
    const bool more = (ks + TK) < K;
    if (more) stage(buf ^ 1, ks + TK);                 // prefetch next slab
    if (more) asm volatile("s_wait_asynccnt 0x4" ::: "memory");  // oldest 4 done
    else      asm volatile("s_wait_asynccnt 0x0" ::: "memory");
    __syncthreads();                                   // slab visible to all waves

    const float* a0p = &atile[buf][(w >> 1) * 32 + lp][8 * hi];
    const v16h a0 = load_afrag_lds(a0p);
    const v16h a1 = load_afrag_lds(a0p + 16 * TK);     // +16 rows
#pragma unroll
    for (int t = 0; t < 4; t++) {
      const _Float16* br = bbase + (size_t)(t * 16) * K + ks;
      v8h g0 = *(const v8h*)(br);
      v8h g1 = *(const v8h*)(br + 16);
      v16h bfr;
#pragma unroll
      for (int q = 0; q < 8; q++) { bfr[q] = g0[q]; bfr[8 + q] = g1[q]; }
      acc[0][t] = __builtin_amdgcn_wmma_f32_16x16x32_f16(false, a0, false, bfr,
                                                         (short)0, acc[0][t], false, false);
      acc[1][t] = __builtin_amdgcn_wmma_f32_16x16x32_f16(false, a1, false, bfr,
                                                         (short)0, acc[1][t], false, false);
    }
    __syncthreads();                                   // all reads done before overwrite
    buf ^= 1;
  }
  // epilogue: bias + act + residual
#pragma unroll
  for (int mi = 0; mi < 2; mi++) {
#pragma unroll
    for (int t = 0; t < 4; t++) {
      const int n  = n0 + t * 16 + lp;
      const float bv = bias ? bias[n] : 0.0f;
#pragma unroll
      for (int r = 0; r < 8; r++) {
        const int row = m0 + mi * 16 + r + 8 * hi;
        float v = act_apply(acc[mi][t][r] + bv, act);
        if (residual) v += residual[(size_t)row * N + n];
        Cout[(size_t)row * N + n] = v;
      }
    }
  }
}

// ---------------------------------------------------------------------------
// fused channel attention gate: g = sigmoid(relu(xl@ca1+b1)@ca2+b2)
// (66 MFLOP total -- kept off the WMMA pipe so all real GEMMs are guard-free)
__global__ void cagate_kernel(const float* __restrict__ xl, const float* __restrict__ ca1w,
                              const float* __restrict__ ca1b, const float* __restrict__ ca2w,
                              const float* __restrict__ ca2b, float* __restrict__ gbuf) {
  __shared__ float xrow[256];
  __shared__ float trow[16];
  const int row = blockIdx.x;
  xrow[threadIdx.x] = xl[(size_t)row * 256 + threadIdx.x];
  __syncthreads();
  if (threadIdx.x < 16) {
    float acc = ca1b[threadIdx.x];
    for (int k = 0; k < 256; k++) acc += xrow[k] * ca1w[k * 16 + threadIdx.x];
    trow[threadIdx.x] = fmaxf(acc, 0.0f);
  }
  __syncthreads();
  float acc = ca2b[threadIdx.x];
#pragma unroll
  for (int d = 0; d < 16; d++) acc += trow[d] * ca2w[d * 256 + threadIdx.x];
  gbuf[(size_t)row * 256 + threadIdx.x] = 1.0f / (1.0f + __expf(-acc));
}

// ---------------------------------------------------------------------------
// node-weight mixing: mixed = softmax(node_w/0.01) . [m0, xo0..xo_{L-1}, o1, g*xl, xl, o4]
__global__ void mix_kernel(const float* __restrict__ m0, const float* __restrict__ x0,
                           const float* __restrict__ x1, const float* __restrict__ x2,
                           const float* __restrict__ x3, const float* __restrict__ o1,
                           const float* __restrict__ g, const float* __restrict__ xl,
                           const float* __restrict__ o4, const float* __restrict__ node_w,
                           int layer, float* __restrict__ outp) {
  int e = blockIdx.x * 256 + threadIdx.x;
  if (e >= MROWS * CDIM) return;
  const int cnt = 5 + layer;
  float nw[8];
  float mx = -1e30f;
  for (int q = 0; q < cnt; q++) { nw[q] = node_w[layer * 8 + q]; mx = fmaxf(mx, nw[q]); }
  float s = 0.0f;
  for (int q = 0; q < cnt; q++) { nw[q] = __expf((nw[q] - mx) * 100.0f); s += nw[q]; }
  const float inv = 1.0f / s;
  const float* xs[4] = {x0, x1, x2, x3};
  float acc = nw[0] * m0[e];
  for (int p = 0; p < layer; p++) acc += nw[1 + p] * xs[p][e];
  const float xlv = xl[e];
  acc += nw[layer + 1] * o1[e];
  acc += nw[layer + 2] * (g[e] * xlv);
  acc += nw[layer + 3] * xlv;
  acc += nw[layer + 4] * o4[e];
  outp[e] = acc * inv;
}

// ---------------------------------------------------------------------------
// per-head QK layernorm + f16 pack:
//   qh/kh : [b,h,n,32] f16 (d=16 real, upper 16 zero-pad for K=32 WMMA)
//   vh    : [b,h,n,16] f16
__global__ void qkprep_kernel(const float* __restrict__ qkv,
                              const float* __restrict__ nqg, const float* __restrict__ nqb,
                              const float* __restrict__ nkg, const float* __restrict__ nkb,
                              int j, _Float16* __restrict__ qh, _Float16* __restrict__ kh,
                              _Float16* __restrict__ vh) {
  int t = blockIdx.x * 256 + threadIdx.x;           // (b, n, head)
  if (t >= BSZ * NMASK * 16) return;
  const int head = t & 15;
  const int n    = (t >> 4) & 1023;
  const int b    = t >> 14;
  const float* base = qkv + (size_t)(b * NMASK + n) * 768;
  float qv[16], kv[16];
  float qmu = 0.f, kmu = 0.f;
#pragma unroll
  for (int d = 0; d < 16; d++) {
    qv[d] = base[head * 16 + d];        qmu += qv[d];
    kv[d] = base[256 + head * 16 + d];  kmu += kv[d];
  }
  qmu *= (1.0f / 16.0f); kmu *= (1.0f / 16.0f);
  float qvr = 0.f, kvr = 0.f;
#pragma unroll
  for (int d = 0; d < 16; d++) {
    float dq = qv[d] - qmu; qvr += dq * dq;
    float dk = kv[d] - kmu; kvr += dk * dk;
  }
  const float qrs = rsqrtf(qvr * (1.0f / 16.0f) + 1e-5f);
  const float krs = rsqrtf(kvr * (1.0f / 16.0f) + 1e-5f);
  const size_t o32 = (((size_t)(b * 16 + head)) * NMASK + n) * 32;
  const size_t o16 = (((size_t)(b * 16 + head)) * NMASK + n) * 16;
#pragma unroll
  for (int d = 0; d < 16; d++) {
    qh[o32 + d]      = (_Float16)((qv[d] - qmu) * qrs * nqg[j * 16 + d] + nqb[j * 16 + d]);
    qh[o32 + 16 + d] = (_Float16)0.0f;
    kh[o32 + d]      = (_Float16)((kv[d] - kmu) * krs * nkg[j * 16 + d] + nkb[j * 16 + d]);
    kh[o32 + 16 + d] = (_Float16)0.0f;
    vh[o16 + d]      = (_Float16)base[512 + head * 16 + d];
  }
}

// ---------------------------------------------------------------------------
// flash attention: wave = one (b,head, 16-query tile). Online softmax,
// S and P@V both on WMMA (K padded to 32). attnout f32 [b, n, head*16+d].
__global__ void flash_attn_kernel(const _Float16* __restrict__ qh,
                                  const _Float16* __restrict__ kh,
                                  const _Float16* __restrict__ vh,
                                  float* __restrict__ attnout) {
  __shared__ _Float16 plds[8][16][16];
  const int lane = threadIdx.x & 31;
  const int w    = threadIdx.x >> 5;
  const int hi   = lane >> 4;
  const int lp   = lane & 15;
  const int task = blockIdx.x * 8 + w;     // 64 bh * 64 qtiles = 4096 tasks
  const int bh   = task >> 6;
  const int qt   = task & 63;
  const int b    = bh >> 4;
  const int head = bh & 15;

  // Q fragment (loaded once)
  v16h aq;
  {
    const _Float16* qr = qh + (((size_t)bh * NMASK) + qt * 16 + lp) * 32 + 8 * hi;
#pragma unroll
    for (int q = 0; q < 8; q++) { aq[q] = qr[q]; aq[8 + q] = qr[16 + q]; }
  }
  v8f O;
  float mrow[8], lrow[8];
#pragma unroll
  for (int r = 0; r < 8; r++) { O[r] = 0.0f; mrow[r] = -1e30f; lrow[r] = 0.0f; }

  v8f zacc;
#pragma unroll
  for (int r = 0; r < 8; r++) zacc[r] = 0.0f;

  for (int kt = 0; kt < 64; kt++) {
    // K fragment: b[q] = K[key = kt*16 + lp][8*hi + q], b[8+q] = K[..][16 + 8*hi + q]   (BT == K row-major)
    v16h bk;
    {
      const _Float16* kr = kh + (((size_t)bh * NMASK) + kt * 16 + lp) * 32 + 8 * hi;
#pragma unroll
      for (int q = 0; q < 8; q++) { bk[q] = kr[q]; bk[8 + q] = kr[16 + q]; }
    }
    v8f s = __builtin_amdgcn_wmma_f32_16x16x32_f16(false, aq, false, bk,
                                                   (short)0, zacc, false, false);
    // online softmax per row (row lives across a 16-lane half)
#pragma unroll
    for (int r = 0; r < 8; r++) {
      float sv = s[r] * 0.25f;                       // / sqrt(16)
      float rm = sv;
#pragma unroll
      for (int msk = 1; msk < 16; msk <<= 1) rm = fmaxf(rm, __shfl_xor(rm, msk, 32));
      const float mn = fmaxf(mrow[r], rm);
      const float sc = __expf(mrow[r] - mn);
      const float p  = __expf(sv - mn);
      float rs = p;
#pragma unroll
      for (int msk = 1; msk < 16; msk <<= 1) rs += __shfl_xor(rs, msk, 32);
      lrow[r] = lrow[r] * sc + rs;
      mrow[r] = mn;
      O[r] *= sc;
      plds[w][r + 8 * hi][lp] = (_Float16)p;         // C-layout -> LDS
    }
    __syncthreads();
    // P fragment from LDS (A layout, K 16..31 zero-pad)
    v16h ap;
#pragma unroll
    for (int q = 0; q < 16; q++) ap[q] = (_Float16)0.0f;
    {
      const _Float16* pr = &plds[w][lp][8 * hi];
#pragma unroll
      for (int q = 0; q < 8; q++) ap[q] = pr[q];
    }
    // V fragment: rows 0..15 real keys, 16..31 zero
    v16h bv;
#pragma unroll
    for (int q = 0; q < 16; q++) bv[q] = (_Float16)0.0f;
    {
      const _Float16* vb = vh + ((size_t)bh * NMASK + kt * 16 + 8 * hi) * 16 + lp;
#pragma unroll
      for (int q = 0; q < 8; q++) bv[q] = vb[q * 16];
    }
    O = __builtin_amdgcn_wmma_f32_16x16x32_f16(false, ap, false, bv,
                                               (short)0, O, false, false);
  }
#pragma unroll
  for (int r = 0; r < 8; r++) {
    const int row = b * NMASK + qt * 16 + r + 8 * hi;
    attnout[(size_t)row * CDIM + head * 16 + lp] = O[r] / lrow[r];
  }
}

// ---------------------------------------------------------------------------
// layernorm over last dim D (block per row)
__global__ void ln_kernel(const float* __restrict__ in, const float* __restrict__ g,
                          const float* __restrict__ bta, float* __restrict__ outp, int D) {
  __shared__ float red[256];
  const int row = blockIdx.x;
  const float* x = in + (size_t)row * D;
  float s = 0.0f;
  for (int i = threadIdx.x; i < D; i += 256) s += x[i];
  red[threadIdx.x] = s; __syncthreads();
  for (int st = 128; st > 0; st >>= 1) {
    if (threadIdx.x < st) red[threadIdx.x] += red[threadIdx.x + st];
    __syncthreads();
  }
  const float mu = red[0] / (float)D;
  __syncthreads();
  float v = 0.0f;
  for (int i = threadIdx.x; i < D; i += 256) { float d = x[i] - mu; v += d * d; }
  red[threadIdx.x] = v; __syncthreads();
  for (int st = 128; st > 0; st >>= 1) {
    if (threadIdx.x < st) red[threadIdx.x] += red[threadIdx.x + st];
    __syncthreads();
  }
  const float rs = rsqrtf(red[0] / (float)D + 1e-5f);
  for (int i = threadIdx.x; i < D; i += 256)
    outp[(size_t)row * D + i] = (x[i] - mu) * rs * g[i] + bta[i];
}

// ---------------------------------------------------------------------------
extern "C" void kernel_launch(void* const* d_in, const int* in_sizes, int n_in,
                              void* d_out, int out_size, void* d_ws, size_t ws_size,
                              hipStream_t stream) {
  (void)in_sizes; (void)n_in; (void)out_size; (void)ws_size;
  // inputs in setup_inputs() order
  const float* x      = (const float*)d_in[0];
  const int*   midx   = (const int*)  d_in[1];
  const float* Wp     = (const float*)d_in[2];
  const float* bp     = (const float*)d_in[3];
  const float* cls    = (const float*)d_in[4];
  const float* qkv_w  = (const float*)d_in[5];
  const float* qkv_b  = (const float*)d_in[6];
  const float* proj_w = (const float*)d_in[7];
  const float* proj_b = (const float*)d_in[8];
  const float* nq_g   = (const float*)d_in[9];
  const float* nq_b   = (const float*)d_in[10];
  const float* nk_g   = (const float*)d_in[11];
  const float* nk_b   = (const float*)d_in[12];
  const float* conv_w = (const float*)d_in[13];
  const float* conv_b = (const float*)d_in[14];
  const float* ca1_w  = (const float*)d_in[15];
  const float* ca1_b  = (const float*)d_in[16];
  const float* ca2_w  = (const float*)d_in[17];
  const float* ca2_b  = (const float*)d_in[18];
  const float* mlp1_w = (const float*)d_in[19];
  const float* mlp1_b = (const float*)d_in[20];
  const float* mlp2_w = (const float*)d_in[21];
  const float* mlp2_b = (const float*)d_in[22];
  const float* node_w = (const float*)d_in[23];
  const float* ln0_g  = (const float*)d_in[24];
  const float* ln0_b  = (const float*)d_in[25];
  const float* h1_w   = (const float*)d_in[26];
  const float* h1_b   = (const float*)d_in[27];
  const float* ln1_g  = (const float*)d_in[28];
  const float* ln1_b  = (const float*)d_in[29];
  const float* h2_w   = (const float*)d_in[30];
  const float* h2_b   = (const float*)d_in[31];
  const float* ln2_g  = (const float*)d_in[32];
  const float* ln2_b  = (const float*)d_in[33];
  const float* h3_w   = (const float*)d_in[34];
  const float* h3_b   = (const float*)d_in[35];
  const float* pr_w   = (const float*)d_in[36];
  const float* pr_b   = (const float*)d_in[37];
  float* out = (float*)d_out;

  // ---- workspace bump allocator
  char* ws = (char*)d_ws;
  size_t off = 0;
  auto carve = [&](size_t bytes) -> void* {
    void* p = ws + off;
    off += (bytes + 255) & ~(size_t)255;
    return p;
  };
  _Float16 *wqkvT[4], *wprojT[4], *wconvT[4], *wmlp1T[4], *wmlp2T[4];
  for (int j = 0; j < 4; j++) {
    wqkvT[j]  = (_Float16*)carve((size_t)768 * 256 * 2);
    wprojT[j] = (_Float16*)carve((size_t)256 * 256 * 2);
    wconvT[j] = (_Float16*)carve((size_t)256 * 256 * 2);
    wmlp1T[j] = (_Float16*)carve((size_t)1024 * 256 * 2);
    wmlp2T[j] = (_Float16*)carve((size_t)256 * 1024 * 2);
  }
  _Float16* wh1T = (_Float16*)carve((size_t)2048 * 256 * 2);
  _Float16* wh2T = (_Float16*)carve((size_t)256 * 2048 * 2);
  _Float16* wh3T = (_Float16*)carve((size_t)2048 * 256 * 2);
  _Float16* wprT = (_Float16*)carve((size_t)512 * 2048 * 2);

  float* m0      = (float*)carve((size_t)MROWS * 256 * 4);
  float* xo[4];
  for (int j = 0; j < 4; j++) xo[j] = (float*)carve((size_t)MROWS * 256 * 4);
  float* o1      = (float*)carve((size_t)MROWS * 256 * 4);
  float* gbuf    = (float*)carve((size_t)MROWS * 256 * 4);
  float* o4      = (float*)carve((size_t)MROWS * 256 * 4);
  float* ubuf    = (float*)carve((size_t)MROWS * 1024 * 4);
  float* mixed   = (float*)carve((size_t)MROWS * 256 * 4);
  float* qkvb    = (float*)carve((size_t)MROWS * 768 * 4);
  float* attnout = (float*)carve((size_t)MROWS * 256 * 4);
  _Float16* qhb  = (_Float16*)carve((size_t)64 * NMASK * 32 * 2);
  _Float16* khb  = (_Float16*)carve((size_t)64 * NMASK * 32 * 2);
  _Float16* vhb  = (_Float16*)carve((size_t)64 * NMASK * 16 * 2);
  float* s1      = (float*)carve((size_t)MROWS * 256 * 4);
  float* s2      = (float*)carve((size_t)MROWS * 256 * 4);
  float* hA      = (float*)carve((size_t)MROWS * 2048 * 4);
  float* hB      = (float*)carve((size_t)MROWS * 2048 * 4);

  auto wcv = [&](const float* W, _Float16* WT, int K, int N) {
    int tot = K * N;
    wconv_kernel<<<(tot + 255) / 256, 256, 0, stream>>>(W, WT, K, N);
  };
  auto gemm = [&](const float* A, const _Float16* BT, const float* bias,
                  const float* resid, float* C, int N, int K, int act) {
    dim3 g(N / 128, MROWS / 128);
    gemm_f16w<<<g, 256, 0, stream>>>(A, BT, bias, resid, C, N, K, act);
  };

  // ---- convert + transpose all WMMA weights to f16 [N,K]
  for (int j = 0; j < 4; j++) {
    wcv(qkv_w  + (size_t)j * 256 * 768,  wqkvT[j],  256, 768);
    wcv(proj_w + (size_t)j * 256 * 256,  wprojT[j], 256, 256);
    wcv(conv_w + (size_t)j * 256 * 256,  wconvT[j], 256, 256);
    wcv(mlp1_w + (size_t)j * 256 * 1024, wmlp1T[j], 256, 1024);
    wcv(mlp2_w + (size_t)j * 1024 * 256, wmlp2T[j], 1024, 256);
  }
  wcv(h1_w, wh1T, 256, 2048);
  wcv(h2_w, wh2T, 2048, 256);
  wcv(h3_w, wh3T, 256, 2048);
  wcv(pr_w, wprT, 2048, 512);

  // ---- patch embed + pos + gather
  patch_gather_kernel<<<(BSZ * NMASK * CDIM + 255) / 256, 256, 0, stream>>>(
      x, midx, Wp, bp, cls, m0);

  // ---- 4 layers
  const float* xl = m0;
  for (int j = 0; j < 4; j++) {
    gemm(xl, wconvT[j], conv_b + j * 256, nullptr, o1, 256, 256, 1);        // gelu
    cagate_kernel<<<MROWS, 256, 0, stream>>>(xl, ca1_w + (size_t)j * 256 * 16,
                                             ca1_b + j * 16, ca2_w + (size_t)j * 16 * 256,
                                             ca2_b + j * 256, gbuf);
    gemm(xl, wmlp1T[j], mlp1_b + j * 1024, nullptr, ubuf, 1024, 256, 1);    // gelu
    gemm(ubuf, wmlp2T[j], mlp2_b + j * 256, nullptr, o4, 256, 1024, 0);
    mix_kernel<<<(MROWS * CDIM + 255) / 256, 256, 0, stream>>>(
        m0, xo[0], xo[1], xo[2], xo[3], o1, gbuf, xl, o4, node_w, j, mixed);
    gemm(mixed, wqkvT[j], qkv_b + j * 768, nullptr, qkvb, 768, 256, 0);
    qkprep_kernel<<<(BSZ * NMASK * 16 + 255) / 256, 256, 0, stream>>>(
        qkvb, nq_g, nq_b, nk_g, nk_b, j, qhb, khb, vhb);
    flash_attn_kernel<<<512, 256, 0, stream>>>(qhb, khb, vhb, attnout);
    gemm(attnout, wprojT[j], proj_b + j * 256, mixed, xo[j], 256, 256, 0);  // +residual
    xl = xo[j];
  }

  // ---- predict head
  ln_kernel<<<MROWS, 256, 0, stream>>>(xo[3], ln0_g, ln0_b, s1, 256);
  gemm(s1, wh1T, h1_b, nullptr, hA, 2048, 256, 0);
  ln_kernel<<<MROWS, 256, 0, stream>>>(hA, ln1_g, ln1_b, hB, 2048);
  gemm(hB, wh2T, h2_b, nullptr, s2, 256, 2048, 0);
  ln_kernel<<<MROWS, 256, 0, stream>>>(s2, ln2_g, ln2_b, s1, 256);
  gemm(s1, wh3T, h3_b, nullptr, hA, 2048, 256, 0);
  gemm(hA, wprT, pr_b, nullptr, out, 512, 2048, 0);
}
